// UnifiedFCQuantumHybrid_65481071402624
// MI455X (gfx1250) — compile-verified, matrix operand order checked by reading, not run on hardware
//
#include <hip/hip_runtime.h>
#include <hip/hip_bf16.h>
#include <math.h>

typedef __attribute__((ext_vector_type(16))) _Float16 v16h;
typedef __attribute__((ext_vector_type(8)))  _Float16 v8h;
typedef __attribute__((ext_vector_type(8)))  float    v8f;

#define BATCH 8192
#define KPAD 800          // 784 + bias col (784) + zero pad, multiple of 32
#define KC2 96            // conv2 im2col K: 72 + bias col (72) padded to 96
#define TWO_PI_F 6.2831853071795864769f

// ---------------- workspace layout (bytes) ----------------
// [0,102400)              : fc1_w as f16 [64][800]  (col784 = fc1_b, rest pad 0)
// [102400,104448)         : fc2_w as f16 [16][64]   (rows 4..15 = 0)
// [104448,107520)         : conv2_w as f16 [16][96] (col72 = conv2_b, rest pad 0)
// [107520,107552)         : BN accumulators, 8 floats
// [107776, +13107200)     : feats f16 [8192][800] (col784 = 1.0)
// [13214976, +1048576)    : h1 f16 [8192][64]
// [14263552, +131072)     : logits f32 [8192][4]

// ------------------------------------------------------------------
// Kernel 1: weight conversion + bias folding + BN accumulator reset
// ------------------------------------------------------------------
__global__ __launch_bounds__(256) void k_prep(const float* __restrict__ fc1_w,
                                              const float* __restrict__ fc1_b,
                                              const float* __restrict__ fc2_w,
                                              const float* __restrict__ conv2_w,
                                              const float* __restrict__ conv2_b,
                                              _Float16* __restrict__ w1h,
                                              _Float16* __restrict__ w2h,
                                              _Float16* __restrict__ w2c,
                                              float* __restrict__ accum) {
  int i = blockIdx.x * 256 + threadIdx.x;
  if (i < 64 * KPAD) {
    int n = i / KPAD, k = i % KPAD;
    float v = (k < 784) ? fc1_w[n * 784 + k] : (k == 784 ? fc1_b[n] : 0.0f);
    w1h[i] = (_Float16)v;
  }
  if (i < 16 * 64) {
    int r = i / 64, c = i % 64;
    w2h[i] = (_Float16)((r < 4) ? fc2_w[r * 64 + c] : 0.0f);
  }
  if (i < 16 * KC2) {
    int oc = i / KC2, k = i % KC2;
    float v = (k < 72) ? conv2_w[oc * 72 + k] : (k == 72 ? conv2_b[oc] : 0.0f);
    w2c[i] = (_Float16)v;
  }
  if (i < 8) accum[i] = 0.0f;
}

// ------------------------------------------------------------------
// WMMA fragment loader (ISA 16-bit A/B^T layout, 16x32 K-major rows):
// lane<16 : row base+lane,    K = {k..k+7, k+16..k+23}
// lane>=16: row base+lane-16, K = {k+8..k+15, k+24..k+31}
// works for global or LDS pointers (generic addrspace).
// ------------------------------------------------------------------
__device__ __forceinline__ v16h load_frag(const _Float16* base) {
  v8h lo = *(const v8h*)base;
  v8h hi = *(const v8h*)(base + 16);
  v16h r;
  #pragma unroll
  for (int i = 0; i < 8; ++i) { r[i] = lo[i]; r[8 + i] = hi[i]; }
  return r;
}

// ------------------------------------------------------------------
// Kernel 2: per image — conv1+relu+pool (VALU), then conv2 as implicit
// GEMM on WMMA (im2col in LDS, bias folded as K-column), then pool.
// ------------------------------------------------------------------
__global__ __launch_bounds__(256) void k_conv(const float* __restrict__ x,
                                              const float* __restrict__ w1,
                                              const float* __restrict__ b1,
                                              const _Float16* __restrict__ w2c,
                                              _Float16* __restrict__ feats) {
  __shared__ float    sx[28 * 28];
  __shared__ float    sw1[72];
  __shared__ float    sb1[8];
  __shared__ float    p1[8 * 16 * 16];       // pooled conv1, zero halo
  __shared__ _Float16 sA[208 * KC2];         // im2col: 196 pixels (+pad) x 96
  __shared__ float    c2[208 * 16];          // conv2 raw output [pixel][oc]

  const int tid  = threadIdx.x;
  const int wave = tid >> 5;
  const int lane = tid & 31;

  const float* xb = x + (size_t)blockIdx.x * 784;
  for (int i = tid; i < 784; i += 256) sx[i] = xb[i];
  for (int i = tid; i < 72; i += 256) sw1[i] = w1[i];
  if (tid < 8) sb1[tid] = b1[tid];
  for (int i = tid; i < 8 * 256; i += 256) p1[i] = 0.0f;
  __syncthreads();

  // conv1 (SAME 28x28) + relu + 2x2 pool -> p1[8][14][14] at +1 halo
  for (int t = tid; t < 8 * 14 * 14; t += 256) {
    int c = t / 196, rem = t % 196, py = rem / 14, px = rem % 14;
    float m = -1e30f;
    #pragma unroll
    for (int dy = 0; dy < 2; ++dy)
      #pragma unroll
      for (int dx = 0; dx < 2; ++dx) {
        int oy = 2 * py + dy, ox = 2 * px + dx;
        float acc = sb1[c];
        #pragma unroll
        for (int ky = 0; ky < 3; ++ky)
          #pragma unroll
          for (int kx = 0; kx < 3; ++kx) {
            int iy = oy + ky - 1, ix = ox + kx - 1;
            if (iy >= 0 && iy < 28 && ix >= 0 && ix < 28)
              acc = fmaf(sw1[c * 9 + ky * 3 + kx], sx[iy * 28 + ix], acc);
          }
        m = fmaxf(m, acc);
      }
    p1[c * 256 + (py + 1) * 16 + (px + 1)] = fmaxf(m, 0.0f);
  }
  __syncthreads();

  // im2col: row = oy*14+ox (196 used, 208 padded), col = ic*9+ky*3+kx (72),
  // col 72 = 1.0 (bias), 73..95 = 0. halo in p1 makes SAME padding implicit.
  for (int i = tid; i < 208 * KC2; i += 256) {
    int row = i / KC2, col = i % KC2;
    float v = 0.0f;
    if (row < 196) {
      if (col < 72) {
        int oy = row / 14, ox = row % 14;
        int ic = col / 9, r = col % 9, ky = r / 3, kx = r % 3;
        v = p1[ic * 256 + (oy + ky) * 16 + (ox + kx)];
      } else if (col == 72) {
        v = 1.0f;
      }
    }
    sA[i] = (_Float16)v;
  }
  __syncthreads();

  // B panel (conv2 weights + bias col) once per wave: 3 k-steps
  v16h bf[3];
  {
    const _Float16* pb = w2c + (size_t)(lane & 15) * KC2 + ((lane >> 4) << 3);
    #pragma unroll
    for (int ks = 0; ks < 3; ++ks) bf[ks] = load_frag(pb + 32 * ks);
  }
  // 13 M-tiles of 16 pixels, LDS -> WMMA
  for (int t = wave; t < 13; t += 8) {
    const _Float16* pa = sA + (size_t)(t * 16 + (lane & 15)) * KC2 + ((lane >> 4) << 3);
    v8f c = {};
    #pragma unroll
    for (int ks = 0; ks < 3; ++ks) {
      v16h a = load_frag(pa + 32 * ks);
      c = __builtin_amdgcn_wmma_f32_16x16x32_f16(false, a, false, bf[ks],
                                                 (short)0, c, false, false);
    }
    const int col = lane & 15;
    const int rbase = t * 16 + ((lane >> 4) << 3);
    #pragma unroll
    for (int r = 0; r < 8; ++r) c2[(rbase + r) * 16 + col] = c[r];
  }
  __syncthreads();

  // relu + 2x2 pool -> feats[c*49+y*7+x]; col 784 = 1.0 (fc1 bias), pad 0
  _Float16* fb = feats + (size_t)blockIdx.x * KPAD;
  for (int t = tid; t < KPAD; t += 256) {
    if (t >= 784) { fb[t] = (_Float16)(t == 784 ? 1.0f : 0.0f); continue; }
    int c = t / 49, rem = t % 49, py = rem / 7, px = rem % 7;
    float m = -1e30f;
    #pragma unroll
    for (int dy = 0; dy < 2; ++dy)
      #pragma unroll
      for (int dx = 0; dx < 2; ++dx)
        m = fmaxf(m, c2[((2 * py + dy) * 14 + (2 * px + dx)) * 16 + c]);
    fb[t] = (_Float16)fmaxf(m, 0.0f);
  }
}

// ------------------------------------------------------------------
// Kernel 3: FC1 GEMM  h1[8192,64] = relu(feats * w1h^T)  (bias in K)
// one wave computes a 16x64 strip: A fragment reused across 4 N-tiles.
// ------------------------------------------------------------------
__global__ __launch_bounds__(256) void k_fc1(const _Float16* __restrict__ feats,
                                             const _Float16* __restrict__ w1h,
                                             _Float16* __restrict__ h1) {
  const int wave = threadIdx.x >> 5;
  const int lane = threadIdx.x & 31;
  const int m0 = (blockIdx.x * 8 + wave) * 16;       // 512 strips
  const int half = (lane >> 4) << 3;
  const _Float16* pa = feats + (size_t)(m0 + (lane & 15)) * KPAD + half;
  const _Float16* pb = w1h + (size_t)(lane & 15) * KPAD + half;

  v8f c0 = {}, c1 = {}, c2 = {}, c3 = {};
  for (int k = 0; k < KPAD; k += 32) {
    __builtin_prefetch(pa + k + 64, 0, 1);           // global_prefetch_b8
    v16h a = load_frag(pa + k);
    v16h b0 = load_frag(pb + k);
    v16h b1 = load_frag(pb + 16 * KPAD + k);
    v16h b2 = load_frag(pb + 32 * KPAD + k);
    v16h b3 = load_frag(pb + 48 * KPAD + k);
    c0 = __builtin_amdgcn_wmma_f32_16x16x32_f16(false, a, false, b0, (short)0, c0, false, false);
    c1 = __builtin_amdgcn_wmma_f32_16x16x32_f16(false, a, false, b1, (short)0, c1, false, false);
    c2 = __builtin_amdgcn_wmma_f32_16x16x32_f16(false, a, false, b2, (short)0, c2, false, false);
    c3 = __builtin_amdgcn_wmma_f32_16x16x32_f16(false, a, false, b3, (short)0, c3, false, false);
  }
  const int col = lane & 15;
  const int rbase = m0 + ((lane >> 4) << 3);
  #pragma unroll
  for (int r = 0; r < 8; ++r) {
    h1[(size_t)(rbase + r) * 64 + col]      = (_Float16)fmaxf(c0[r], 0.0f);
    h1[(size_t)(rbase + r) * 64 + col + 16] = (_Float16)fmaxf(c1[r], 0.0f);
    h1[(size_t)(rbase + r) * 64 + col + 32] = (_Float16)fmaxf(c2[r], 0.0f);
    h1[(size_t)(rbase + r) * 64 + col + 48] = (_Float16)fmaxf(c3[r], 0.0f);
  }
}

// ------------------------------------------------------------------
// Kernel 4: FC2 (N padded 4->16) + BN partial sums via atomics.
// ------------------------------------------------------------------
__global__ __launch_bounds__(256) void k_fc2(const _Float16* __restrict__ h1,
                                             const _Float16* __restrict__ w2h,
                                             const float* __restrict__ b2,
                                             float* __restrict__ logits,
                                             float* __restrict__ accum) {
  const int wave = threadIdx.x >> 5;
  const int lane = threadIdx.x & 31;
  const int m0 = (blockIdx.x * 8 + wave) * 16;       // 512 tiles
  const int half = (lane >> 4) << 3;
  const _Float16* pa = h1 + (size_t)(m0 + (lane & 15)) * 64 + half;
  const _Float16* pb = w2h + (size_t)(lane & 15) * 64 + half;

  v8f c = {};
  #pragma unroll
  for (int k = 0; k < 64; k += 32) {
    v16h a = load_frag(pa + k);
    v16h b = load_frag(pb + k);
    c = __builtin_amdgcn_wmma_f32_16x16x32_f16(false, a, false, b,
                                               (short)0, c, false, false);
  }
  const int col = lane & 15;
  const int rbase = m0 + ((lane >> 4) << 3);
  if (col < 4) {
    const float bias = b2[col];
    float s = 0.0f, ss = 0.0f;
    #pragma unroll
    for (int r = 0; r < 8; ++r) {
      float v = c[r] + bias;
      logits[(size_t)(rbase + r) * 4 + col] = v;
      s += v;
      ss += v * v;
    }
    atomicAdd(&accum[col], s);
    atomicAdd(&accum[4 + col], ss);
  }
}

// ------------------------------------------------------------------
// Kernel 5: BatchNorm + angle map + 4-qubit statevector sim + add.
// qubit q <-> bit (3-q) of the flattened [2,2,2,2] index.
// ------------------------------------------------------------------
struct Cplx { float re, im; };
__device__ __forceinline__ Cplx cmul(Cplx a, Cplx b) {
  return {a.re * b.re - a.im * b.im, a.re * b.im + a.im * b.re};
}
__device__ __forceinline__ Cplx cadd(Cplx a, Cplx b) { return {a.re + b.re, a.im + b.im}; }

__device__ __forceinline__ void apply1q(Cplx* psi, int q, Cplx u00, Cplx u01,
                                        Cplx u10, Cplx u11) {
  const int m = 1 << (3 - q);
  #pragma unroll
  for (int idx = 0; idx < 16; ++idx) {
    if (idx & m) continue;
    Cplx a0 = psi[idx], a1 = psi[idx | m];
    psi[idx]     = cadd(cmul(u00, a0), cmul(u01, a1));
    psi[idx | m] = cadd(cmul(u10, a0), cmul(u11, a1));
  }
}

__global__ __launch_bounds__(256) void k_final(const float* __restrict__ logits,
                                               const float* __restrict__ accum,
                                               const float* __restrict__ gamma,
                                               const float* __restrict__ beta,
                                               const float* __restrict__ var_p,
                                               float* __restrict__ out) {
  const int b = blockIdx.x * 256 + threadIdx.x;
  const float invB = 1.0f / (float)BATCH;

  float cl[4];
  #pragma unroll
  for (int j = 0; j < 4; ++j) {
    float mu = accum[j] * invB;
    float var = accum[4 + j] * invB - mu * mu;
    cl[j] = (logits[(size_t)b * 4 + j] - mu) * rsqrtf(var + 1e-5f) * gamma[j] + beta[j];
  }
  float mn = fminf(fminf(cl[0], cl[1]), fminf(cl[2], cl[3]));
  float mx = fmaxf(fmaxf(cl[0], cl[1]), fmaxf(cl[2], cl[3]));
  float ang[4];
  #pragma unroll
  for (int j = 0; j < 4; ++j)
    ang[j] = (cl[j] - mn) / (mx - mn + 1e-6f) * TWO_PI_F;

  Cplx psi[16];
  #pragma unroll
  for (int i = 0; i < 16; ++i) psi[i] = {0.0f, 0.0f};
  psi[0] = {1.0f, 0.0f};

  #pragma unroll
  for (int q = 0; q < 4; ++q) {                       // Z feature map
    float ch = __cosf(0.5f * ang[q]), sh = __sinf(0.5f * ang[q]);
    apply1q(psi, q, {ch, -sh}, {0, 0}, {0, 0}, {ch, sh});
  }
  #pragma unroll
  for (int q = 0; q < 4; ++q) {                       // RX/RY/RZ ansatz
    float tx = var_p[3 * q], ty = var_p[3 * q + 1], tz = var_p[3 * q + 2];
    float cx = __cosf(0.5f * tx), sx = __sinf(0.5f * tx);
    apply1q(psi, q, {cx, 0}, {0, -sx}, {0, -sx}, {cx, 0});
    float cy = __cosf(0.5f * ty), sy = __sinf(0.5f * ty);
    apply1q(psi, q, {cy, 0}, {-sy, 0}, {sy, 0}, {cy, 0});
    float cz = __cosf(0.5f * tz), sz = __sinf(0.5f * tz);
    apply1q(psi, q, {cz, -sz}, {0, 0}, {0, 0}, {cz, sz});
  }
  #pragma unroll
  for (int q = 0; q < 3; ++q) {                       // CNOT chain
    const int mc = 1 << (3 - q), mt = 1 << (3 - (q + 1));
    #pragma unroll
    for (int idx = 0; idx < 16; ++idx)
      if ((idx & mc) && !(idx & mt)) {
        Cplx t = psi[idx]; psi[idx] = psi[idx | mt]; psi[idx | mt] = t;
      }
  }
  float p[16];
  #pragma unroll
  for (int i = 0; i < 16; ++i) p[i] = psi[i].re * psi[i].re + psi[i].im * psi[i].im;

  float expz[4];
  #pragma unroll
  for (int k = 0; k < 4; ++k) {
    const int m = 1 << (3 - k);
    float e = 0.0f;
    #pragma unroll
    for (int i = 0; i < 16; ++i) e += (i & m) ? -p[i] : p[i];
    expz[k] = e;
  }
  #pragma unroll
  for (int j = 0; j < 4; ++j)
    out[(size_t)b * 4 + j] = cl[j] + expz[3 - j];     // Pauli label reversal
}

// ------------------------------------------------------------------
extern "C" void kernel_launch(void* const* d_in, const int* in_sizes, int n_in,
                              void* d_out, int out_size, void* d_ws, size_t ws_size,
                              hipStream_t stream) {
  const float* x       = (const float*)d_in[0];
  const float* conv1_w = (const float*)d_in[1];
  const float* conv1_b = (const float*)d_in[2];
  const float* conv2_w = (const float*)d_in[3];
  const float* conv2_b = (const float*)d_in[4];
  const float* fc1_w   = (const float*)d_in[5];
  const float* fc1_b   = (const float*)d_in[6];
  const float* fc2_w   = (const float*)d_in[7];
  const float* fc2_b   = (const float*)d_in[8];
  const float* bn_g    = (const float*)d_in[9];
  const float* bn_b    = (const float*)d_in[10];
  const float* var_p   = (const float*)d_in[11];
  float* out = (float*)d_out;

  char* ws = (char*)d_ws;
  _Float16* w1h   = (_Float16*)(ws);                  // 102400 B
  _Float16* w2h   = (_Float16*)(ws + 102400);         //   2048 B
  _Float16* w2c   = (_Float16*)(ws + 104448);         //   3072 B
  float*    accum = (float*)   (ws + 107520);         //     32 B
  _Float16* feats = (_Float16*)(ws + 107776);         // 13107200 B
  _Float16* h1    = (_Float16*)(ws + 13214976);       // 1048576 B
  float*    lg    = (float*)   (ws + 14263552);       // 131072 B

  k_prep <<<200,  256, 0, stream>>>(fc1_w, fc1_b, fc2_w, conv2_w, conv2_b,
                                    w1h, w2h, w2c, accum);
  k_conv <<<BATCH,256, 0, stream>>>(x, conv1_w, conv1_b, w2c, feats);
  k_fc1  <<<64,   256, 0, stream>>>(feats, w1h, h1);
  k_fc2  <<<64,   256, 0, stream>>>(h1, w2h, fc2_b, lg, accum);
  k_final<<<BATCH/256, 256, 0, stream>>>(lg, accum, bn_g, bn_b, var_p, out);
}
